// BipartiteGNN_71124658422090
// MI455X (gfx1250) — matrix operand order repaired and checked by reading, not compiled, not run on hardware
//
#include <hip/hip_runtime.h>

typedef __attribute__((ext_vector_type(16))) _Float16 v16h;
typedef __attribute__((ext_vector_type(8)))  _Float16 v8h;
typedef __attribute__((ext_vector_type(8)))  float    v8f;

#define WAVES 8
#define TPB   (WAVES * 32)

// ---------------------------------------------------------------------------
// A-fragment loader for V_WMMA_F32_16X16X32_F16 (16x32 f16 A matrix).
// Lane m (m<16) holds row m, K = {0..15}; lane m+16 holds row m, K = {16..31}
// (contiguous order, matching the B fragments).
// => two contiguous 16-byte chunks at half-offsets hi*16 and hi*16+8.
// ---------------------------------------------------------------------------
__device__ __forceinline__ v16h load_a_frag(const _Float16* __restrict__ row, int hi) {
  const v8h a0 = *(const v8h*)(row + hi * 16);
  const v8h a1 = *(const v8h*)(row + hi * 16 + 8);
  return __builtin_shufflevector(a0, a1, 0, 1, 2, 3, 4, 5, 6, 7,
                                         8, 9, 10, 11, 12, 13, 14, 15);
}

// B-fragment (32x16 f16 B tile) from fp32 weights W[K][32] row-major.
// Lane n (n<16) holds col n, K=0..15; lane n+16 holds col n, K=16..31.
__device__ __forceinline__ v16h load_b_frag(const float* __restrict__ W,
                                            int kbase, int nbase, int lane) {
  const int n  = (lane & 15) + nbase;
  const int kb = kbase + ((lane >> 4) << 4);
  v16h b;
#pragma unroll
  for (int i = 0; i < 16; ++i) b[i] = (_Float16)W[(kb + i) * 32 + n];
  return b;
}

// B-fragment matching the permuted H layout used for layer 2.
// H column k is stored at half-slot s(k) = 2*(k&15) + (k>>4); the fragment
// slot p = 16*half + element therefore carries H column k(p) = (p&1)*16 + (p>>1), so the weight
// row loaded for slot p must be k(p). Contraction order over K is irrelevant.
__device__ __forceinline__ v16h load_b_frag_perm(const float* __restrict__ W,
                                                 int nbase, int lane) {
  const int n  = (lane & 15) + nbase;
  const int kb = (lane >> 4) << 4;
  v16h b;
#pragma unroll
  for (int i = 0; i < 16; ++i) {
    const int s = kb + i;                     // fragment slot: 16*half + element
    const int k = ((s & 1) << 4) + (s >> 1);  // actual H column in that slot
    b[i] = (_Float16)W[k * 32 + n];
  }
  return b;
}

#define WMMA_F16(A, B, C)                                                     \
  __builtin_amdgcn_wmma_f32_16x16x32_f16(false, (A), false, (B), (short)0,    \
                                         (C), false, false)

// WGP-scope prefetch (scope 0 is the default encoding and has no named
// assembler token): pulls the line into all cache levels, unlike the
// SE/SYS-scope prefetches which stop at GL2 (useless here: L2-resident data).
__device__ __forceinline__ void prefetch_wgp(const void* p) {
  asm volatile("global_prefetch_b8 %0, off" :: "v"(p));
}

// ---------------------------------------------------------------------------
// Edge MLP + scatter:  j = relu(relu([A[eCons], V[eVars]] @ Wj1 + bj1) @ Wj2
//                               + bj2);  outSum[scatter_idx[e]] += j
// One 16-edge tile per wave per iteration; 6 WMMAs per tile.
// Indices are preloaded 2 iterations ahead; gather rows for the next
// iteration are warmed with WGP-scope prefetches (indices already resident).
// ---------------------------------------------------------------------------
__global__ void __launch_bounds__(TPB)
edge_mlp_scatter(const _Float16* __restrict__ Ah,   // [*,32] f16 (c or out_c)
                 const _Float16* __restrict__ Vh,   // [NV,32] f16
                 const int* __restrict__ eCons,
                 const int* __restrict__ eVars,
                 const int scatSel,                 // 0: scatter eCons, 1: eVars
                 const float* __restrict__ Wj1, const float* __restrict__ bj1,
                 const float* __restrict__ Wj2, const float* __restrict__ bj2,
                 float* __restrict__ outSum, long long E)
{
  // Per-wave H tile, permuted columns: uint index c holds H cols (c, c+16).
  __shared__ __attribute__((aligned(16))) unsigned ldsH[WAVES][16][16];
  const int lane = threadIdx.x & 31;
  const int w    = threadIdx.x >> 5;
  const int hi   = lane >> 4;
  const int col  = lane & 15;

  // Persistent weight fragments (tiny, hoisted out of the edge loop).
  const v16h b1a0 = load_b_frag(Wj1,  0,  0, lane);   // Wj1 rows 0..31,  cols 0..15
  const v16h b1b0 = load_b_frag(Wj1, 32,  0, lane);   // Wj1 rows 32..63, cols 0..15
  const v16h b1a1 = load_b_frag(Wj1,  0, 16, lane);
  const v16h b1b1 = load_b_frag(Wj1, 32, 16, lane);
  const v16h b2_0 = load_b_frag_perm(Wj2,  0, lane);
  const v16h b2_1 = load_b_frag_perm(Wj2, 16, lane);
  const float bj1_0 = bj1[col], bj1_1 = bj1[col + 16];
  const float bj2_0 = bj2[col], bj2_1 = bj2[col + 16];

  const long long nTiles = (E + 15) >> 4;
  long long tile = (long long)blockIdx.x * WAVES + w;         // wave-uniform
  const long long tstr   = (long long)gridDim.x * WAVES;
  const long long tstr16 = tstr << 4;

  auto clampE = [&](long long x) { return x >= E ? E - 1 : x; };

  long long base = tile << 4;
  // Index pipeline: iteration i uses (ca0,va0); (ca1,va1) is i+1 (resident),
  // and each iteration issues loads for i+2.
  const long long g0 = clampE(base + col);
  int ca0 = eCons[g0], va0 = eVars[g0];
  const long long g1 = clampE(base + tstr16 + col);
  int ca1 = eCons[g1], va1 = eVars[g1];

  for (; tile < nTiles; tile += tstr, base += tstr16) {
    // Issue index loads for iteration i+2 (awaited only next iteration).
    const long long g2 = clampE(base + 2 * tstr16 + col);
    const int ca2 = eCons[g2];
    const int va2 = eVars[g2];

    // Warm next iteration's gather rows into WGP$/L0 (indices already resident).
    prefetch_wgp(Ah + (long long)ca1 * 32);
    prefetch_wgp(Vh + (long long)va1 * 32);

    // Gather X = [c-part | v-part] directly in WMMA A layout (2x16B per lane).
    const v16h xa = load_a_frag(Ah + (long long)ca0 * 32, hi);
    const v16h xb = load_a_frag(Vh + (long long)va0 * 32, hi);

    // Layer 1: H = relu(X @ Wj1 + b1), K=64 as two K=32 chunks, N=32 as 2 tiles.
    v8f acc0 = {}, acc1 = {};
    acc0 = WMMA_F16(xa, b1a0, acc0);
    acc0 = WMMA_F16(xb, b1b0, acc0);
    acc1 = WMMA_F16(xa, b1a1, acc1);
    acc1 = WMMA_F16(xb, b1b1, acc1);

    // Bias + ReLU; stash H in LDS (permuted cols -> one ds_store_b32 per row).
#pragma unroll
    for (int r = 0; r < 8; ++r) {
      const int m = r + 8 * hi;               // row index of this D element
      union { _Float16 h[2]; unsigned u; } pk;
      pk.h[0] = (_Float16)fmaxf(acc0[r] + bj1_0, 0.0f);   // col
      pk.h[1] = (_Float16)fmaxf(acc1[r] + bj1_1, 0.0f);   // col+16
      ldsH[w][m][col] = pk.u;
    }
    // DS ops are in-order within a wave: stores above complete before loads.
    const v16h ha = load_a_frag((const _Float16*)&ldsH[w][col][0], hi);

    // Layer 2: J = relu(H @ Wj2 + b2) with permutation-matched B fragments.
    v8f o0 = {}, o1 = {};
    o0 = WMMA_F16(ha, b2_0, o0);
    o1 = WMMA_F16(ha, b2_1, o1);

    // Scatter: segment-sum via L2-side fp32 atomic adds (no return).
    const int scat = scatSel ? va0 : ca0;     // this lane's row index (row=col)
#pragma unroll
    for (int r = 0; r < 8; ++r) {
      const int m = r + 8 * hi;
      const int node = __shfl(scat, m, 32);   // full-EXEC shuffle, then guard
      const long long e = base + m;
      const float j0 = fmaxf(o0[r] + bj2_0, 0.0f);
      const float j1 = fmaxf(o1[r] + bj2_1, 0.0f);
      if (e < E) {
        unsafeAtomicAdd(&outSum[(long long)node * 32 + col],      j0);
        unsafeAtomicAdd(&outSum[(long long)node * 32 + col + 16], j1);
      }
    }

    // Rotate the index pipeline.
    ca0 = ca1; va0 = va1;
    ca1 = ca2; va1 = va2;
  }
}

// ---------------------------------------------------------------------------
// Node embedding: out = relu(feat @ W + b); fp32 + f16 copies.
// One thread per (node, dim).
// ---------------------------------------------------------------------------
__global__ void __launch_bounds__(TPB)
embed_nodes(const float* __restrict__ feat, int F,
            const float* __restrict__ W, const float* __restrict__ b,
            float* __restrict__ outF, _Float16* __restrict__ outH, long long N)
{
  const long long total = N * 32;
  for (long long gid = (long long)blockIdx.x * blockDim.x + threadIdx.x;
       gid < total; gid += (long long)gridDim.x * blockDim.x) {
    const long long node = gid >> 5;
    const int d = (int)(gid & 31);
    float acc = b[d];
    for (int k = 0; k < F; ++k) acc += feat[node * F + k] * W[k * 32 + d];
    const float r = fmaxf(acc, 0.0f);
    outF[gid] = r;
    outH[gid] = (_Float16)r;
  }
}

// out_c = relu([sum, base] @ W + b) -> f16 only (consumed as WMMA gather input)
__global__ void __launch_bounds__(TPB)
node_repr_h(const float* __restrict__ sum, const float* __restrict__ base,
            const float* __restrict__ W, const float* __restrict__ b,
            _Float16* __restrict__ outH, long long N)
{
  const long long total = N * 32;
  for (long long gid = (long long)blockIdx.x * blockDim.x + threadIdx.x;
       gid < total; gid += (long long)gridDim.x * blockDim.x) {
    const long long node = gid >> 5;
    const int d = (int)(gid & 31);
    const float* s = sum + node * 32;
    const float* c = base + node * 32;
    float acc = b[d];
#pragma unroll
    for (int k = 0; k < 32; ++k) acc += s[k] * W[k * 32 + d];
#pragma unroll
    for (int k = 0; k < 32; ++k) acc += c[k] * W[(32 + k) * 32 + d];
    outH[gid] = (_Float16)fmaxf(acc, 0.0f);
  }
}

// ---------------------------------------------------------------------------
// out_v = relu([vsSum, v] @ Wvr + bvr); h = relu(relu(out_v@Wo1+bo1)@Wo2+bo2);
// out = h @ Wo3 + bo3.  One wave per node, LDS for inter-layer broadcast
// (wave-internal, DS in-order, no barriers).
// ---------------------------------------------------------------------------
__global__ void __launch_bounds__(TPB)
final_out(const float* __restrict__ vsSum, const float* __restrict__ v,
          const float* __restrict__ Wvr, const float* __restrict__ bvr,
          const float* __restrict__ Wo1, const float* __restrict__ bo1,
          const float* __restrict__ Wo2, const float* __restrict__ bo2,
          const float* __restrict__ Wo3, const float* __restrict__ bo3,
          float* __restrict__ out, long long NV)
{
  __shared__ float buf[WAVES][32];
  const int lane = threadIdx.x & 31;
  const int w    = threadIdx.x >> 5;
  const long long nstr = (long long)gridDim.x * WAVES;
  for (long long node = (long long)blockIdx.x * WAVES + w; node < NV; node += nstr) {
    const float* s  = vsSum + node * 32;
    const float* vv = v + node * 32;
    float acc = bvr[lane];
#pragma unroll
    for (int k = 0; k < 32; ++k) acc += s[k] * Wvr[k * 32 + lane];
#pragma unroll
    for (int k = 0; k < 32; ++k) acc += vv[k] * Wvr[(32 + k) * 32 + lane];
    buf[w][lane] = fmaxf(acc, 0.0f);

    float a1 = bo1[lane];
#pragma unroll
    for (int k = 0; k < 32; ++k) a1 += buf[w][k] * Wo1[k * 32 + lane];
    buf[w][lane] = fmaxf(a1, 0.0f);   // all reads above issue before this store

    float a2 = bo2[lane];
#pragma unroll
    for (int k = 0; k < 32; ++k) a2 += buf[w][k] * Wo2[k * 32 + lane];
    a2 = fmaxf(a2, 0.0f);

    float p = a2 * Wo3[lane];
#pragma unroll
    for (int o = 16; o > 0; o >>= 1) p += __shfl_down(p, o, 32);
    if (lane == 0) out[node] = p + bo3[0];
  }
}

// ---------------------------------------------------------------------------
extern "C" void kernel_launch(void* const* d_in, const int* in_sizes, int n_in,
                              void* d_out, int out_size, void* d_ws, size_t ws_size,
                              hipStream_t stream)
{
  const float* consF = (const float*)d_in[0];
  const float* varsF = (const float*)d_in[1];
  const int*   eIdx  = (const int*)d_in[2];
  const float* Wc  = (const float*)d_in[3];  const float* bc  = (const float*)d_in[4];
  const float* Wv  = (const float*)d_in[5];  const float* bv  = (const float*)d_in[6];
  const float* Wj1 = (const float*)d_in[7];  const float* bj1 = (const float*)d_in[8];
  const float* Wj2 = (const float*)d_in[9];  const float* bj2 = (const float*)d_in[10];
  const float* Wcr = (const float*)d_in[11]; const float* bcr = (const float*)d_in[12];
  const float* Wvr = (const float*)d_in[13]; const float* bvr = (const float*)d_in[14];
  const float* Wo1 = (const float*)d_in[15]; const float* bo1 = (const float*)d_in[16];
  const float* Wo2 = (const float*)d_in[17]; const float* bo2 = (const float*)d_in[18];
  const float* Wo3 = (const float*)d_in[19]; const float* bo3 = (const float*)d_in[20];

  const long long NC = in_sizes[0] / 2;   // CF = 2
  const long long NV = in_sizes[1] / 9;   // VF = 9
  const long long E  = in_sizes[2] / 2;
  const int* eCons = eIdx;
  const int* eVars = eIdx + E;

  // Workspace partition (~70 MB total)
  char* ws = (char*)d_ws;
  size_t off = 0;
  auto carve = [&](size_t bytes) {
    void* p = ws + off;
    off += (bytes + 255) & ~(size_t)255;
    return p;
  };
  float*    c     = (float*)   carve((size_t)NC * 32 * 4);
  float*    v     = (float*)   carve((size_t)NV * 32 * 4);
  _Float16* cH    = (_Float16*)carve((size_t)NC * 32 * 2);
  _Float16* vH    = (_Float16*)carve((size_t)NV * 32 * 2);
  _Float16* ocH   = (_Float16*)carve((size_t)NC * 32 * 2);
  float*    csSum = (float*)   carve((size_t)NC * 32 * 4);
  float*    vsSum = (float*)   carve((size_t)NV * 32 * 4);
  (void)ws_size; (void)n_in; (void)out_size;

  const int nodeBlocksC = (int)((NC * 32 + TPB - 1) / TPB);
  const int nodeBlocksV = (int)((NV * 32 + TPB - 1) / TPB);
  const int edgeBlocks  = 2048;   // 16384 waves over E/16 = 200k tiles
  const int finBlocks   = (int)((NV + WAVES - 1) / WAVES);

  // Node embeddings
  embed_nodes<<<nodeBlocksC, TPB, 0, stream>>>(consF, 2, Wc, bc, c, cH, NC);
  embed_nodes<<<nodeBlocksV, TPB, 0, stream>>>(varsF, 9, Wv, bv, v, vH, NV);

  // Pass 1: edges -> constraints
  (void)hipMemsetAsync(csSum, 0, (size_t)NC * 32 * 4, stream);
  edge_mlp_scatter<<<edgeBlocks, TPB, 0, stream>>>(
      cH, vH, eCons, eVars, /*scatSel=*/0, Wj1, bj1, Wj2, bj2, csSum, E);
  node_repr_h<<<nodeBlocksC, TPB, 0, stream>>>(csSum, c, Wcr, bcr, ocH, NC);

  // Pass 2: edges -> variables
  (void)hipMemsetAsync(vsSum, 0, (size_t)NV * 32 * 4, stream);
  edge_mlp_scatter<<<edgeBlocks, TPB, 0, stream>>>(
      ocH, vH, eCons, eVars, /*scatSel=*/1, Wj1, bj1, Wj2, bj2, vsSum, E);

  // out_v representation + output MLP head
  final_out<<<finBlocks, TPB, 0, stream>>>(vsSum, v, Wvr, bvr, Wo1, bo1,
                                           Wo2, bo2, Wo3, bo3,
                                           (float*)d_out, NV);
}
